// Codebook_64063732187187
// MI455X (gfx1250) — matrix operand order repaired and checked by reading, not compiled, not run on hardware
//
#include <hip/hip_runtime.h>

// ---------------------------------------------------------------------------
// VQ nearest-neighbor (argmin_k ||z - c_k||^2) via bf16x3-split WMMA GEMM.
//   dist2(n,k) ~ c_sq[k] - 2 * dot(z_n, c_k)      (z_sq constant per row)
// dot computed as hi*hi + hi*lo + lo*hi in V_WMMA_F32_16X16X32_BF16 (fp32 acc),
// which recovers ~fp32 accuracy. Output vectors gathered from fp32 codebook.
// ---------------------------------------------------------------------------

typedef __attribute__((ext_vector_type(16))) __bf16 bf16x16;
typedef __attribute__((ext_vector_type(8)))  float  v8f;

#define NROWS          65536   // 32 * 2048
#define DIM            64
#define KCODES         1024
#define CHUNK          128     // codebook rows staged in LDS per iteration
#define ROWS_PER_BLOCK 128     // 8 waves * 16 rows
#define LDS_STRIDE     72      // bf16 elems per code row (64 + 8 pad), 144B:
                               // 8B-aligned for b64 LDS reads, breaks the
                               // 128B power-of-two bank-conflict stride

union BFrag {
    unsigned long long q[4];
    bf16x16 v;
};

// 32B B-operand fragment from LDS via 4x 8-byte loads (8B-aligned addresses).
__device__ inline bf16x16 load_b_frag(const __bf16* p) {
    BFrag u;
    const unsigned long long* s = reinterpret_cast<const unsigned long long*>(p);
    u.q[0] = s[0]; u.q[1] = s[1]; u.q[2] = s[2]; u.q[3] = s[3];
    return u.v;
}

__global__ __launch_bounds__(256)
void vq_nearest_kernel(const float* __restrict__ z,
                       const float* __restrict__ cb,
                       float* __restrict__ out)
{
    __shared__ __bf16 sHi[CHUNK * LDS_STRIDE];   // 18 KB
    __shared__ __bf16 sLo[CHUNK * LDS_STRIDE];   // 18 KB
    __shared__ float  sCsq[CHUNK];
    __shared__ int    sIdx[ROWS_PER_BLOCK];

    const int tid  = threadIdx.x;
    const int lane = tid & 31;          // wave32
    const int wave = tid >> 5;          // 8 waves / block
    const int half = lane >> 4;         // 0: lanes 0-15, 1: lanes 16-31
    const int l15  = lane & 15;

    const int rowBase = blockIdx.x * ROWS_PER_BLOCK + wave * 16;

    // ---- A fragments: this wave's 16x64 z tile, bf16 hi + lo --------------
    // A 16x32 k order used here (contiguous, matches B): lane (0-15) holds row M=lane;
    // lanes 16-31 hold row M=lane-16 with K offset +16.
    // elems 0..15 -> K = kOff+0..15,
    // kOff = half*16 (+32 for the second K-chunk).
    bf16x16 aHi[2], aLo[2];
    {
        const int myRow = rowBase + l15;
        #pragma unroll
        for (int kc = 0; kc < 2; ++kc) {
            const float* zr = z + myRow * DIM + kc * 32 + half * 16;
            #pragma unroll
            for (int e = 0; e < 16; ++e) {
                const int k = e;   // 0..15
                const float f = zr[k];
                const __bf16 h = (__bf16)f;
                aHi[kc][e] = h;
                aLo[kc][e] = (__bf16)(f - (float)h);
            }
        }
    }

    // Running (min, argmin). D layout: lane's v8f elem r covers output row
    // M = half*8 + r, column N = l15 (the candidate code).
    float best[8];
    int   bidx[8];
    #pragma unroll
    for (int r = 0; r < 8; ++r) { best[r] = 3.4e38f; bidx[r] = 0; }

    for (int chunk = 0; chunk < KCODES; chunk += CHUNK) {
        __syncthreads();   // previous chunk's LDS fully consumed

        // ---- stage 128 codebook rows -> bf16 hi/lo in LDS + fp32 c_sq ----
        if (tid < CHUNK) {
            const float* cr = cb + (chunk + tid) * DIM;
            float ss = 0.0f;
            #pragma unroll 8
            for (int d = 0; d < DIM; ++d) {
                const float f = cr[d];
                ss += f * f;
                const __bf16 h = (__bf16)f;
                sHi[tid * LDS_STRIDE + d] = h;
                sLo[tid * LDS_STRIDE + d] = (__bf16)(f - (float)h);
            }
            sCsq[tid] = ss;
            // Pull the next chunk toward L2/L0 while we compute this one.
            if (chunk + CHUNK < KCODES)
                __builtin_prefetch(cb + (chunk + CHUNK + tid) * DIM, 0, 0);
        }
        __syncthreads();

        // ---- 8 tiles of 16 codes; 2 K-chunks x 3 split-WMMAs each --------
        #pragma unroll
        for (int tile = 0; tile < CHUNK / 16; ++tile) {
            const int codeCol = tile * 16 + l15;   // B column N = l15
            v8f acc = {};
            #pragma unroll
            for (int kc = 0; kc < 2; ++kc) {
                // B k order used here (contiguous, same as A): lanes 0-15 hold K=0..15, lanes 16-31 hold
                // K=16..31 (two bf16 per VGPR) -> 16 contiguous bf16 in LDS.
                const int off = codeCol * LDS_STRIDE + half * 16 + kc * 32;
                const bf16x16 bHi = load_b_frag(&sHi[off]);
                const bf16x16 bLo = load_b_frag(&sLo[off]);
                acc = __builtin_amdgcn_wmma_f32_16x16x32_bf16(
                        false, aHi[kc], false, bHi, (short)0, acc, false, false);
                acc = __builtin_amdgcn_wmma_f32_16x16x32_bf16(
                        false, aHi[kc], false, bLo, (short)0, acc, false, false);
                acc = __builtin_amdgcn_wmma_f32_16x16x32_bf16(
                        false, aLo[kc], false, bHi, (short)0, acc, false, false);
            }
            const float csq = sCsq[tile * 16 + l15];
            const int   code = chunk + tile * 16 + l15;
            #pragma unroll
            for (int r = 0; r < 8; ++r) {
                const float d2 = csq - 2.0f * acc[r];
                if (d2 < best[r]) { best[r] = d2; bidx[r] = code; }
            }
        }
    }

    // ---- reduce (min, idx) across the 16 candidate columns per row --------
    // xor masks < 16 stay within each 16-lane half.
    #pragma unroll
    for (int s = 1; s < 16; s <<= 1) {
        #pragma unroll
        for (int r = 0; r < 8; ++r) {
            const float ob = __shfl_xor(best[r], s, 32);
            const int   oi = __shfl_xor(bidx[r], s, 32);
            if (ob < best[r] || (ob == best[r] && oi < bidx[r])) {
                best[r] = ob; bidx[r] = oi;
            }
        }
    }
    if (l15 == 0) {                     // lanes 0 and 16 hold final results
        #pragma unroll
        for (int r = 0; r < 8; ++r)
            sIdx[wave * 16 + half * 8 + r] = bidx[r];
    }
    __syncthreads();

    // ---- gather: out[row] = codebook[argmin] in exact fp32, coalesced -----
    const float4* cb4  = reinterpret_cast<const float4*>(cb);
    float4*       out4 = reinterpret_cast<float4*>(out) +
                         (size_t)blockIdx.x * ROWS_PER_BLOCK * (DIM / 4);
    #pragma unroll
    for (int e = tid; e < ROWS_PER_BLOCK * (DIM / 4); e += 256) {
        const int row  = e >> 4;
        const int comp = e & 15;
        out4[row * (DIM / 4) + comp] = cb4[sIdx[row] * (DIM / 4) + comp];
    }
}

extern "C" void kernel_launch(void* const* d_in, const int* in_sizes, int n_in,
                              void* d_out, int out_size, void* d_ws, size_t ws_size,
                              hipStream_t stream) {
    const float* z  = (const float*)d_in[0];   // [65536, 64] fp32
    const float* cb = (const float*)d_in[1];   // [1024, 64] fp32
    float* out = (float*)d_out;                // [65536, 64] fp32
    (void)in_sizes; (void)n_in; (void)out_size; (void)d_ws; (void)ws_size;

    dim3 grid(NROWS / ROWS_PER_BLOCK);         // 512 blocks
    dim3 block(256);                           // 8 wave32 per block
    vq_nearest_kernel<<<grid, block, 0, stream>>>(z, cb, out);
}
